// SHGNN_55035710931174
// MI455X (gfx1250) — matrix operand order repaired and checked by reading, not compiled, not run on hardware
//
#include <hip/hip_runtime.h>
#include <hip/hip_bf16.h>

// SHGNN pipeline for MI455X (gfx1250, wave32).
// Dense GEMMs run on v_wmma_f32_16x16x32_bf16 (f32 accum); sparse stages
// (GCN scatter, segment softmax/mean) use f32 atomics. Memory-bound overall,
// so gathered incidence features are stored bf16 to halve the big streams.
// GEMM M-dims are padded to multiples of 16 (pad rows zeroed) so the WMMA
// inner loop has unconditional b128 loads/stores and no EXEC manipulation.

typedef __bf16 bf16;
typedef __attribute__((ext_vector_type(16))) __bf16 v16bf;
typedef __attribute__((ext_vector_type(8)))  float  v8f;
typedef __attribute__((ext_vector_type(4)))  unsigned uint4v;

union TileHalfU { uint4v u[2]; v16bf bf; };

#define TPB 256

#define N_NODES 50000
#define N_EDGES 25000
#define P_E     400000
#define M_E     800000
#define P_N     400000
#define M_N     800000
#define FEAT    128
#define DIM     128
#define HEADS   4
#define DH      32
#define NCLS    40

static inline unsigned cdiv(long long a, long long b) { return (unsigned)((a + b - 1) / b); }

// ---------- monotone float<->uint mapping for atomicMax on floats ----------
__device__ __forceinline__ unsigned fkey(float f) {
  unsigned u = __float_as_uint(f);
  return (u & 0x80000000u) ? ~u : (u | 0x80000000u);
}
__device__ __forceinline__ float funkey(unsigned m) {
  return __uint_as_float((m & 0x80000000u) ? (m ^ 0x80000000u) : ~m);
}

// ---------------------------- elementwise utils ----------------------------
__global__ void k_fill(float* __restrict__ p, float v, long long n) {
  long long i = (long long)blockIdx.x * blockDim.x + threadIdx.x;
  if (i < n) p[i] = v;
}

__global__ void k_f32_to_bf16(const float* __restrict__ s, bf16* __restrict__ d, long long n) {
  long long i = (long long)blockIdx.x * blockDim.x + threadIdx.x;
  if (i < n) d[i] = (bf16)s[i];
}

// out[i,:] = (bf16) x[map[i],:] ; one thread handles 8 contiguous columns
__global__ void k_gather_bf16(const float* __restrict__ x, const int* __restrict__ map,
                              bf16* __restrict__ out, int rows, int cols) {
  int cb = cols >> 3;                                   // 8-col chunks per row
  long long t = (long long)blockIdx.x * blockDim.x + threadIdx.x;
  if (t >= (long long)rows * cb) return;
  int i = (int)(t / cb), c8 = (int)(t % cb);
  const float* xp = x + (size_t)map[i] * cols + (size_t)c8 * 8;
  union { uint4v u; bf16 b[8]; } o;
#pragma unroll
  for (int e = 0; e < 8; ++e) o.b[e] = (bf16)xp[e];
  *(uint4v*)(out + (size_t)i * cols + (size_t)c8 * 8) = o.u;
}

// --------------------------- WMMA bf16 GEMMs (N=128) ------------------------
// One wave computes one 16x16 f32 tile of C = A[M,K] @ B[K,128]; 8 waves per
// block cover the full N=128 stripe of one 16-row tile. Requires M % 16 == 0
// (pad rows of A zeroed by caller). K is a compile-time constant: the loop
// fully unrolls into b128 load clauses + back-to-back v_wmma.
// A layout (16x32 bf16, ISA 7.12.2): lane<16 -> K {0..7,16..23}, lane>=16 ->
// K {8..15,24..31}; each half is two contiguous 16B chunks -> 2x b128 loads.
template <int K>
__global__ __launch_bounds__(TPB)
void k_gemm3_wmma(const bf16* __restrict__ A,
                  const bf16* __restrict__ B0, const bf16* __restrict__ B1,
                  const bf16* __restrict__ B2,
                  float* __restrict__ C0, float* __restrict__ C1, float* __restrict__ C2) {
  const int N = 128;
  int lane = threadIdx.x & 31;
  int nt   = threadIdx.x >> 5;        // 0..7 -> n tile
  int m0   = blockIdx.x * 16;
  int n0   = nt * 16;
  int half = lane >> 4;               // K half of the wave
  int lm   = lane & 15;
  size_t abase = (size_t)(m0 + lm) * K + half * 8;
  size_t bbase = (size_t)(half * 8) * N + n0 + lm;
  v8f acc0 = {}, acc1 = {}, acc2 = {};
#pragma unroll
  for (int k0 = 0; k0 < K; k0 += 32) {
    TileHalfU a, b0, b1, b2;
    const uint4v* ap  = (const uint4v*)(A + abase + k0);
    const uint4v* bp0 = (const uint4v*)(B0 + bbase + (size_t)k0 * N);
    const uint4v* bp1 = (const uint4v*)(B1 + bbase + (size_t)k0 * N);
    const uint4v* bp2 = (const uint4v*)(B2 + bbase + (size_t)k0 * N);
    a.u[0]  = ap[0];  a.u[1]  = ap[2];      // K +0..7, +16..23 (16B each)
#pragma unroll
    for (int s = 0; s < 16; ++s) {
      size_t ko = (size_t)(s < 8 ? s : s + 8) * N;
      b0.bf[s] = B0[bbase + (size_t)k0 * N + ko];
      b1.bf[s] = B1[bbase + (size_t)k0 * N + ko];
      b2.bf[s] = B2[bbase + (size_t)k0 * N + ko];
    }
    acc0 = __builtin_amdgcn_wmma_f32_16x16x32_bf16(false, a.bf, false, b0.bf, (short)0, acc0, false, false);
    acc1 = __builtin_amdgcn_wmma_f32_16x16x32_bf16(false, a.bf, false, b1.bf, (short)0, acc1, false, false);
    acc2 = __builtin_amdgcn_wmma_f32_16x16x32_bf16(false, a.bf, false, b2.bf, (short)0, acc2, false, false);
  }
#pragma unroll
  for (int r = 0; r < 8; ++r) {
    size_t o = (size_t)(m0 + r + 8 * half) * N + n0 + lm;   // M = vgpr + 8*(lane/16)
    C0[o] = acc0[r]; C1[o] = acc1[r]; C2[o] = acc2[r];
  }
}

__global__ __launch_bounds__(TPB)
void k_gemm1_wmma(const bf16* __restrict__ A, const bf16* __restrict__ B,
                  float* __restrict__ C) {   // K = N = 128, M % 16 == 0
  const int K = 128, N = 128;
  int lane = threadIdx.x & 31;
  int nt   = threadIdx.x >> 5;
  int m0   = blockIdx.x * 16;
  int n0   = nt * 16;
  int half = lane >> 4;
  int lm   = lane & 15;
  size_t abase = (size_t)(m0 + lm) * K + half * 8;
  size_t bbase = (size_t)(half * 8) * N + n0 + lm;
  v8f acc = {};
#pragma unroll
  for (int k0 = 0; k0 < K; k0 += 32) {
    TileHalfU a, b;
    const uint4v* ap = (const uint4v*)(A + abase + k0);
    const uint4v* bp = (const uint4v*)(B + bbase + (size_t)k0 * N);
    a.u[0] = ap[0]; a.u[1] = ap[2];
#pragma unroll
    for (int s = 0; s < 16; ++s)
      b.bf[s] = B[bbase + (size_t)k0 * N + (size_t)(s < 8 ? s : s + 8) * N];
    acc = __builtin_amdgcn_wmma_f32_16x16x32_bf16(false, a.bf, false, b.bf, (short)0, acc, false, false);
  }
#pragma unroll
  for (int r = 0; r < 8; ++r)
    C[(size_t)(m0 + r + 8 * half) * N + n0 + lm] = acc[r];
}

// ------------------------------- GCN pieces --------------------------------
__global__ void k_deg_accum(const int* __restrict__ dst, int m, float* __restrict__ deg) {
  long long e = (long long)blockIdx.x * blockDim.x + threadIdx.x;
  if (e < m) atomicAdd(&deg[dst[e]], 1.0f);
}

__global__ void k_rsqrt_inplace(float* __restrict__ p, int n) {
  long long i = (long long)blockIdx.x * blockDim.x + threadIdx.x;
  if (i < n) p[i] = rsqrtf(fmaxf(p[i], 1.0f));   // deg>=1 due to self loop
}

// out[i,c] = h[i,c]*dinv[i]^2 + bias[c]   (self-loop term + bias init)
__global__ void k_gcn_self(const float* __restrict__ h, const float* __restrict__ dinv,
                           const float* __restrict__ bias, float* __restrict__ out, int rows) {
  long long idx = (long long)blockIdx.x * blockDim.x + threadIdx.x;
  if (idx >= (long long)rows * DIM) return;
  int i = (int)(idx >> 7), c = (int)(idx & 127);
  float di = dinv[i];
  out[idx] = h[idx] * di * di + bias[c];
}

__global__ void k_gcn_scatter(const float* __restrict__ h, const float* __restrict__ dinv,
                              const int* __restrict__ src, const int* __restrict__ dst,
                              float* __restrict__ out, int m) {
  long long idx = (long long)blockIdx.x * blockDim.x + threadIdx.x;
  if (idx >= (long long)m * DIM) return;
  int e = (int)(idx >> 7), c = (int)(idx & 127);
  int s = src[e], d = dst[e];
  float nrm = dinv[s] * dinv[d];
  atomicAdd(&out[(size_t)d * DIM + c], h[(size_t)s * DIM + c] * nrm);
}

// ----------------------------- segment mean --------------------------------
__global__ void k_cnt_accum(const int* __restrict__ seg, int rows, float* __restrict__ cnt) {
  long long i = (long long)blockIdx.x * blockDim.x + threadIdx.x;
  if (i < rows) atomicAdd(&cnt[seg[i]], 1.0f);
}

__global__ void k_seg_accum(const float* __restrict__ x, const int* __restrict__ seg,
                            float* __restrict__ sum, int rows) {
  long long idx = (long long)blockIdx.x * blockDim.x + threadIdx.x;
  if (idx >= (long long)rows * DIM) return;
  int i = (int)(idx >> 7), c = (int)(idx & 127);
  atomicAdd(&sum[(size_t)seg[i] * DIM + c], x[idx]);
}

__global__ void k_seg_div(float* __restrict__ sum, const float* __restrict__ cnt, int S) {
  long long idx = (long long)blockIdx.x * blockDim.x + threadIdx.x;
  if (idx >= (long long)S * DIM) return;
  sum[idx] = sum[idx] / fmaxf(cnt[idx >> 7], 1.0f);
}

// -------------------------------- PMA pool ---------------------------------
__global__ void k_scores(const float* __restrict__ Km, const float* __restrict__ seed,
                         float* __restrict__ sc, int rows) {
  long long idx = (long long)blockIdx.x * blockDim.x + threadIdx.x;
  if (idx >= (long long)rows * HEADS) return;
  int i = (int)(idx >> 2), h = (int)(idx & 3);
  const float* kp = Km + (size_t)i * DIM + h * DH;
  const float* sp = seed + h * DH;
  float s = 0.f;
#pragma unroll
  for (int c = 0; c < DH; ++c) s += kp[c] * sp[c];
  sc[idx] = s * 0.17677669529663687f;  // 1/sqrt(32)
}

__global__ void k_smax(const float* __restrict__ sc, const int* __restrict__ seg,
                       unsigned* __restrict__ smax, int rows) {
  long long idx = (long long)blockIdx.x * blockDim.x + threadIdx.x;
  if (idx >= (long long)rows * HEADS) return;
  int i = (int)(idx >> 2), h = (int)(idx & 3);
  atomicMax(&smax[(size_t)seg[i] * HEADS + h], fkey(sc[idx]));
}

__global__ void k_pma_accum(const float* __restrict__ sc, const unsigned* __restrict__ smax,
                            const float* __restrict__ V, const int* __restrict__ seg,
                            float* __restrict__ den, float* __restrict__ pool, int rows) {
  long long idx = (long long)blockIdx.x * blockDim.x + threadIdx.x;
  if (idx >= (long long)rows * HEADS) return;
  int i = (int)(idx >> 2), h = (int)(idx & 3);
  int s = seg[i];
  float ex = __expf(sc[idx] - funkey(smax[(size_t)s * HEADS + h]));
  atomicAdd(&den[(size_t)s * HEADS + h], ex);
  const float* vp = V + (size_t)i * DIM + h * DH;
  float* pp = pool + (size_t)s * DIM + h * DH;
#pragma unroll
  for (int c = 0; c < DH; ++c) atomicAdd(&pp[c], ex * vp[c]);
}

__global__ void k_pma_norm(const float* __restrict__ pool, const float* __restrict__ den,
                           bf16* __restrict__ out, int S) {
  long long idx = (long long)blockIdx.x * blockDim.x + threadIdx.x;
  if (idx >= (long long)S * DIM) return;
  int s = (int)(idx >> 7), h = (int)((idx & 127) >> 5);
  out[idx] = (bf16)(pool[idx] / fmaxf(den[(size_t)s * HEADS + h], 1e-9f));
}

// ------------------------------ concat + relu ------------------------------
__global__ void k_concat_relu(const float* __restrict__ conv, const float* __restrict__ pma,
                              float* __restrict__ out, int S) {
  long long idx = (long long)blockIdx.x * blockDim.x + threadIdx.x;
  if (idx >= (long long)S * 2 * DIM) return;
  int s = (int)(idx >> 8), c = (int)(idx & 255);
  float v = (c < DIM) ? conv[(size_t)s * DIM + c] : pma[(size_t)s * DIM + (c - DIM)];
  out[idx] = fmaxf(v, 0.0f);
}

// ------------------------------ classifier ---------------------------------
__global__ void k_cls(const float* __restrict__ X, const float* __restrict__ W,
                      const float* __restrict__ b, float* __restrict__ out, int nrows) {
  long long idx = (long long)blockIdx.x * blockDim.x + threadIdx.x;
  if (idx >= (long long)nrows * NCLS) return;
  int n = (int)(idx / NCLS), k = (int)(idx % NCLS);
  const float* xp = X + (size_t)n * (2 * DIM);
  float s = b[k];
#pragma unroll 8
  for (int j = 0; j < 2 * DIM; ++j) s += xp[j] * W[(size_t)j * NCLS + k];
  out[idx] = s;
}

__global__ void k_logsm(float* __restrict__ out, int nrows) {
  long long n = (long long)blockIdx.x * blockDim.x + threadIdx.x;
  if (n >= nrows) return;
  float* r = out + n * NCLS;
  float m = -3.4e38f;
#pragma unroll
  for (int k = 0; k < NCLS; ++k) m = fmaxf(m, r[k]);
  float s = 0.f;
#pragma unroll
  for (int k = 0; k < NCLS; ++k) s += __expf(r[k] - m);
  float l = m + __logf(s);
#pragma unroll
  for (int k = 0; k < NCLS; ++k) r[k] = r[k] - l;
}

// ------------------------- one hypergraph half-step -------------------------
template <int KDIM>
static void run_phase(hipStream_t st,
                      const bf16* A, int rows,
                      const bf16* Wg, const bf16* Wk, const bf16* Wv, const bf16* Wo,
                      const float* gcn_b, const float* seed,
                      const int* conv_src, const int* conv_dst, int m_edges,
                      const int* seg, int S,
                      float* H, float* Kf, float* Vf, float* GCN, float* DEG,
                      float* SEGSUM, float* CNT, float* SC, unsigned* SMAX, float* DEN,
                      float* POOL, bf16* POOLB, float* PMA, float* outcat) {
  // h = A@Wg, K = A@Wk, V = A@Wv  (A reused, 3 WMMAs per A tile; rows%16==0)
  k_gemm3_wmma<KDIM><<<rows / 16, TPB, 0, st>>>(A, Wg, Wk, Wv, H, Kf, Vf);
  // GCN symmetric-norm conv over incidence rows
  k_fill<<<cdiv(rows, TPB), TPB, 0, st>>>(DEG, 1.0f, rows);             // self loops
  k_deg_accum<<<cdiv(m_edges, TPB), TPB, 0, st>>>(conv_dst, m_edges, DEG);
  k_rsqrt_inplace<<<cdiv(rows, TPB), TPB, 0, st>>>(DEG, rows);
  k_gcn_self<<<cdiv((long long)rows * DIM, TPB), TPB, 0, st>>>(H, DEG, gcn_b, GCN, rows);
  k_gcn_scatter<<<cdiv((long long)m_edges * DIM, TPB), TPB, 0, st>>>(H, DEG, conv_src, conv_dst, GCN, m_edges);
  // segment mean -> SEGSUM holds ex_conv
  k_fill<<<cdiv((long long)S * DIM, TPB), TPB, 0, st>>>(SEGSUM, 0.f, (long long)S * DIM);
  k_fill<<<cdiv(S, TPB), TPB, 0, st>>>(CNT, 0.f, S);
  k_cnt_accum<<<cdiv(rows, TPB), TPB, 0, st>>>(seg, rows, CNT);
  k_seg_accum<<<cdiv((long long)rows * DIM, TPB), TPB, 0, st>>>(GCN, seg, SEGSUM, rows);
  k_seg_div<<<cdiv((long long)S * DIM, TPB), TPB, 0, st>>>(SEGSUM, CNT, S);
  // PMA pooled attention
  k_fill<<<cdiv((long long)S * HEADS, TPB), TPB, 0, st>>>((float*)SMAX, 0.f, (long long)S * HEADS); // fkey(-inf)>0
  k_fill<<<cdiv((long long)S * HEADS, TPB), TPB, 0, st>>>(DEN, 0.f, (long long)S * HEADS);
  k_fill<<<cdiv((long long)S * DIM, TPB), TPB, 0, st>>>(POOL, 0.f, (long long)S * DIM);
  k_scores<<<cdiv((long long)rows * HEADS, TPB), TPB, 0, st>>>(Kf, seed, SC, rows);
  k_smax<<<cdiv((long long)rows * HEADS, TPB), TPB, 0, st>>>(SC, seg, SMAX, rows);
  k_pma_accum<<<cdiv((long long)rows * HEADS, TPB), TPB, 0, st>>>(SC, SMAX, Vf, seg, DEN, POOL, rows);
  k_pma_norm<<<cdiv((long long)S * DIM, TPB), TPB, 0, st>>>(POOL, DEN, POOLB, S);
  // pad POOLB rows S..Spad-1 with zeros so the PMA GEMM needs no M guards
  int Spad = (S + 15) & ~15;
  if (Spad > S) {
    long long padf = (long long)(Spad - S) * DIM / 2;   // bf16 pad as f32 words
    k_fill<<<cdiv(padf, TPB), TPB, 0, st>>>((float*)(POOLB + (size_t)S * DIM), 0.f, padf);
  }
  k_gemm1_wmma<<<Spad / 16, TPB, 0, st>>>(POOLB, Wo, PMA);
  // relu(concat(conv, pma))
  k_concat_relu<<<cdiv((long long)S * 2 * DIM, TPB), TPB, 0, st>>>(SEGSUM, PMA, outcat, S);
}

// --------------------------------- driver ----------------------------------
extern "C" void kernel_launch(void* const* d_in, const int* in_sizes, int n_in,
                              void* d_out, int out_size, void* d_ws, size_t ws_size,
                              hipStream_t stream) {
  (void)in_sizes; (void)n_in; (void)out_size;
  const float* node_x   = (const float*)d_in[0];
  const int*   eb_map   = (const int*)d_in[1];
  const int*   eb_ei    = (const int*)d_in[2];
  const int*   eb_batch = (const int*)d_in[3];
  const int*   nb_map   = (const int*)d_in[4];
  const int*   nb_ei    = (const int*)d_in[5];
  const int*   nb_batch = (const int*)d_in[6];
  const float* n2e_W    = (const float*)d_in[7];
  const float* n2e_b    = (const float*)d_in[8];
  const float* n2e_Wk   = (const float*)d_in[9];
  const float* n2e_Wv   = (const float*)d_in[10];
  const float* n2e_seed = (const float*)d_in[11];
  const float* n2e_Wo   = (const float*)d_in[12];
  const float* e2n_W    = (const float*)d_in[13];
  const float* e2n_b    = (const float*)d_in[14];
  const float* e2n_Wk   = (const float*)d_in[15];
  const float* e2n_Wv   = (const float*)d_in[16];
  const float* e2n_seed = (const float*)d_in[17];
  const float* e2n_Wo   = (const float*)d_in[18];
  const float* cls_W    = (const float*)d_in[19];
  const float* cls_b    = (const float*)d_in[20];
  float* out = (float*)d_out;

  // ---- workspace carve-out (256B aligned) ----
  char* base = (char*)d_ws;
  size_t off = 0;
  auto take = [&](size_t bytes) -> char* {
    char* r = base + off;
    off += (bytes + 255) & ~(size_t)255;
    return r;
  };
  bf16*     ABF   = (bf16*)take((size_t)P_N * 2 * DIM * sizeof(bf16));   // gathered features
  float*    H     = (float*)take((size_t)P_N * DIM * sizeof(float));
  float*    Kf    = (float*)take((size_t)P_N * DIM * sizeof(float));
  float*    Vf    = (float*)take((size_t)P_N * DIM * sizeof(float));
  float*    GCN   = (float*)take((size_t)P_N * DIM * sizeof(float));
  float*    DEG   = (float*)take((size_t)P_N * sizeof(float));
  float*    SEGS  = (float*)take((size_t)N_NODES * DIM * sizeof(float));
  float*    CNT   = (float*)take((size_t)N_NODES * sizeof(float));
  float*    SC    = (float*)take((size_t)P_N * HEADS * sizeof(float));
  unsigned* SMAX  = (unsigned*)take((size_t)N_NODES * HEADS * sizeof(unsigned));
  float*    DEN   = (float*)take((size_t)N_NODES * HEADS * sizeof(float));
  float*    POOL  = (float*)take((size_t)N_NODES * DIM * sizeof(float));
  bf16*     POOLB = (bf16*)take((size_t)N_NODES * DIM * sizeof(bf16));   // >= Spad rows
  float*    PMA   = (float*)take((size_t)N_NODES * DIM * sizeof(float)); // >= Spad rows
  float*    EDGEX = (float*)take((size_t)N_EDGES * 2 * DIM * sizeof(float));
  float*    NOUT  = (float*)take((size_t)N_NODES * 2 * DIM * sizeof(float));
  bf16*     WB    = (bf16*)take((size_t)(4 * DIM * DIM + 3 * 2 * DIM * DIM) * sizeof(bf16));
  if (off > ws_size) return;  // insufficient scratch: nothing safe to do

  bf16* n2eWb  = WB;
  bf16* n2eWkb = n2eWb  + DIM * DIM;
  bf16* n2eWvb = n2eWkb + DIM * DIM;
  bf16* n2eWob = n2eWvb + DIM * DIM;
  bf16* e2nWb  = n2eWob + DIM * DIM;
  bf16* e2nWkb = e2nWb  + 2 * DIM * DIM;
  bf16* e2nWvb = e2nWkb + 2 * DIM * DIM;
  bf16* e2nWob = e2nWvb + 2 * DIM * DIM;

  // ---- weights -> bf16 ----
  k_f32_to_bf16<<<cdiv(DIM * DIM, TPB), TPB, 0, stream>>>(n2e_W,  n2eWb,  DIM * DIM);
  k_f32_to_bf16<<<cdiv(DIM * DIM, TPB), TPB, 0, stream>>>(n2e_Wk, n2eWkb, DIM * DIM);
  k_f32_to_bf16<<<cdiv(DIM * DIM, TPB), TPB, 0, stream>>>(n2e_Wv, n2eWvb, DIM * DIM);
  k_f32_to_bf16<<<cdiv(DIM * DIM, TPB), TPB, 0, stream>>>(n2e_Wo, n2eWob, DIM * DIM);
  k_f32_to_bf16<<<cdiv(2 * DIM * DIM, TPB), TPB, 0, stream>>>(e2n_W,  e2nWb,  2 * DIM * DIM);
  k_f32_to_bf16<<<cdiv(2 * DIM * DIM, TPB), TPB, 0, stream>>>(e2n_Wk, e2nWkb, 2 * DIM * DIM);
  k_f32_to_bf16<<<cdiv(2 * DIM * DIM, TPB), TPB, 0, stream>>>(e2n_Wv, e2nWvb, 2 * DIM * DIM);
  k_f32_to_bf16<<<cdiv(DIM * DIM, TPB), TPB, 0, stream>>>(e2n_Wo, e2nWob, DIM * DIM);

  // ---- phase A: node -> edge ----
  k_gather_bf16<<<cdiv((long long)P_E * (FEAT / 8), TPB), TPB, 0, stream>>>(node_x, eb_map, ABF, P_E, FEAT);
  run_phase<FEAT>(stream, ABF, P_E,
                  n2eWb, n2eWkb, n2eWvb, n2eWob, n2e_b, n2e_seed,
                  eb_ei, eb_ei + M_E, M_E, eb_batch, N_EDGES,
                  H, Kf, Vf, GCN, DEG, SEGS, CNT, SC, SMAX, DEN, POOL, POOLB, PMA, EDGEX);

  // ---- phase B: edge -> node ----
  k_gather_bf16<<<cdiv((long long)P_N * (2 * DIM / 8), TPB), TPB, 0, stream>>>(EDGEX, nb_map, ABF, P_N, 2 * DIM);
  run_phase<2 * DIM>(stream, ABF, P_N,
                     e2nWb, e2nWkb, e2nWvb, e2nWob, e2n_b, e2n_seed,
                     nb_ei, nb_ei + M_N, M_N, nb_batch, N_NODES,
                     H, Kf, Vf, GCN, DEG, SEGS, CNT, SC, SMAX, DEN, POOL, POOLB, PMA, NOUT);

  // ---- classifier + log_softmax ----
  k_cls<<<cdiv((long long)N_NODES * NCLS, TPB), TPB, 0, stream>>>(NOUT, cls_W, cls_b, out, N_NODES);
  k_logsm<<<cdiv(N_NODES, TPB), TPB, 0, stream>>>(out, N_NODES);
}
